// EdgeUpdate_7739531067657
// MI455X (gfx1250) — matrix operand order repaired and checked by reading, not compiled, not run on hardware
//
#include <hip/hip_runtime.h>
#include <hip/hip_bf16.h>
#include <math.h>

// ---------------- problem constants ----------------
#define E_TOT   65536
#define N_NODES 8192
#define NGR     16

// repacked weight fragment table (each fragment = 512 bf16 = one 16x16x32-step B tile)
#define FRAG_RH   0      // rbf@w_rh : 4 kt x 4 nt = 16
#define FRAG_BS   16     // Phi_s    : 163 kt x 2 nt = 326
#define FRAG_BVY  342    // Phi_vy   : 65 kt x 2 nt = 130
#define FRAG_BV   472    // Phi_v    : 65 kt x 1 nt = 65  (shared by c=0..2)
#define FRAG_BCR  537    // Phi_cr   : 65 kt x 1 nt = 65  (shared by c=0..2)
#define NFRAG     602

// workspace layout (float indices)
#define WS_SUMS   0      // [16][16]
#define WS_SUMSQ  256    // [16][16]
#define WS_SUMV   512    // [16][8]
#define WS_CNT    640    // [16]
#define WS_MU     656    // [16][16]
#define WS_RMSS   912    // [16]
#define WS_RMSV   928    // [16]
#define WS_SOUT   2048
#define WS_VOUT   (2048 + E_TOT*16)
#define WS_BPK_BYTES ((size_t)(2048 + E_TOT*16 + E_TOT*24) * 4)

typedef __attribute__((ext_vector_type(16))) __bf16        v16bf;
typedef __attribute__((ext_vector_type(8)))  float         v8f;
typedef __attribute__((ext_vector_type(4)))  float         f32x4;
typedef __attribute__((ext_vector_type(4)))  unsigned int  u32x4;

union AFrag { unsigned short us[16]; u32x4 q[2]; v16bf v; };

__device__ __forceinline__ unsigned short f2bf(float f) {
  unsigned u = __builtin_bit_cast(unsigned, f);
  unsigned r = (u + 0x7FFFu + ((u >> 16) & 1u)) >> 16;   // RNE
  return (unsigned short)r;
}

// pack two f32 -> packed bf16x2 (low = a)
__device__ __forceinline__ unsigned pk_bf16(float a, float b) {
#if defined(__AMDGCN__) && __has_builtin(__builtin_amdgcn_cvt_pk_bf16_f32)
  typedef __attribute__((ext_vector_type(2))) __bf16 bf16x2_t;
  bf16x2_t r = __builtin_amdgcn_cvt_pk_bf16_f32(a, b);
  return __builtin_bit_cast(unsigned, r);
#elif defined(__AMDGCN__) && __has_builtin(__builtin_amdgcn_perm)
  // round-half-up to bf16, merge high halves with one v_perm_b32
  unsigned u0 = __builtin_bit_cast(unsigned, a) + 0x8000u;
  unsigned u1 = __builtin_bit_cast(unsigned, b) + 0x8000u;
  return __builtin_amdgcn_perm(u1, u0, 0x07060302u);
#else
  return (unsigned)f2bf(a) | ((unsigned)f2bf(b) << 16);
#endif
}
__device__ __forceinline__ float sigm(float x) { return 1.0f / (1.0f + expf(-x)); }

// ---------------- kernel: zero stats ----------------
__global__ void zero_stats_k(float* ws) {
  int t = threadIdx.x;
  if (t < 1024) ws[t] = 0.0f;
}

// ---------------- kernel: repack w_rh / w_ro / b_ro into bf16 WMMA B fragments ----------------
// B layout per fragment: lane l holds N = l%16, K = 8*(l/16) + (t&7) + 16*(t>>3), t = 0..15.
__global__ void repack_k(const float* __restrict__ w_rh, const float* __restrict__ w_ro,
                         const float* __restrict__ b_ro, unsigned short* __restrict__ Bpk) {
  int tid = blockIdx.x * 256 + threadIdx.x;
  if (tid >= NFRAG * 512) return;
  int f = tid >> 9, r = tid & 511;
  int lane = r >> 4, t = r & 15;
  int Kb  = ((lane >> 4) << 3) + (t & 7) + ((t >> 3) << 4);   // K within 32-chunk
  int n16 = lane & 15;
  float val = 0.0f;
  if (f < FRAG_BS) {                                    // rbf @ w_rh (128x64)
    int g = f - FRAG_RH; int kt = g >> 2, nt = g & 3;
    int K = kt * 32 + Kb, n = nt * 16 + n16;
    val = w_rh[K * 64 + n];
  } else if (f < FRAG_BVY) {                            // Phi_s block: K=65*80 (pad 5216), N=32 (w1|w2)
    int g = f - FRAG_BS; int kt = g >> 1, nt = g & 1;
    int kap = kt * 32 + Kb;
    if (kap < 5200) {
      int k = kap / 80, i = kap - k * 80;
      int o = nt * 16 + n16;
      int col = (o < 24) ? (i * 24 + o) : (1920 + i * 8 + (o - 24));
      float src = (k < 64) ? w_ro[k * 3840 + col] : b_ro[col];
      val = src * 0.1118033988749895f;                  // 1/sqrt(80)
    }
  } else if (f < FRAG_BV) {                             // Phi_vy block: K=65*32, N=24 (w4)
    int g = f - FRAG_BVY; int kt = g >> 1, nt = g & 1;
    int kap = kt * 32 + Kb; int k = kap >> 5, i = kap & 31;
    int o = nt * 16 + n16;
    if (o < 24) {
      int col = 2816 + i * 24 + o;
      float src = (k < 64) ? w_ro[k * 3840 + col] : b_ro[col];
      val = src * 0.10206207261596575f;                 // 1/sqrt(96)
    }
  } else if (f < FRAG_BCR) {                            // Phi_v block: K=65*32, N=8 (w3)
    int kt = f - FRAG_BV;
    int kap = kt * 32 + Kb; int k = kap >> 5, i = kap & 31;
    if (n16 < 8) {
      int col = 2560 + i * 8 + n16;
      float src = (k < 64) ? w_ro[k * 3840 + col] : b_ro[col];
      val = src * 0.17677669529663687f;                 // 1/sqrt(32)
    }
  } else {                                              // Phi_cross block: K=65*32, N=8 (w5)
    int kt = f - FRAG_BCR;
    int kap = kt * 32 + Kb; int k = kap >> 5, i = kap & 31;
    if (n16 < 8) {
      int col = 3584 + i * 8 + n16;
      float src = (k < 64) ? w_ro[k * 3840 + col] : b_ro[col];
      val = src * 0.125f;                               // 1/sqrt(64)
    }
  }
  Bpk[tid] = f2bf(val);
}

// ---------------- WMMA GEMM over rank-1 features Phi[e, k*F+i] = h[e,k]*feat[e,i] ----------------
// F and 32 are both multiples of 8, so each 8-long A-fragment run never crosses an i-wrap:
// k is constant per run, feats are 8 contiguous (32B-aligned) floats. No LDS round trip,
// no division: (k,i) tracked incrementally.
template <int F, int KT, int NT>
__device__ __forceinline__ void gemm_block(int lane,
                                           const float* __restrict__ feat,   // LDS row of this lane's edge
                                           const float* __restrict__ hrow,   // LDS h row: h[64]=1, h[65]=0
                                           const unsigned short* __restrict__ Bpk,
                                           int fragBase, v8f* acc) {
  const int half = lane >> 4;
  const u32x4* __restrict__ bp = (const u32x4*)(Bpk + (size_t)fragBase * 512 + lane * 16);
  int k0 = 0, i00 = half * 8;        // run0: kappa = kt*32 + 8*half      -> (k0, i00)
  for (int kt = 0; kt < KT; ++kt) {
    int i01 = i00 + 16, k1 = k0;     // run1: kappa = kt*32 + 16 + 8*half
    if (i01 >= F) { i01 -= F; k1 += 1; }
    float h0 = hrow[k0], h1 = hrow[k1];
    const f32x4* f0 = (const f32x4*)(feat + i00);
    const f32x4* f1 = (const f32x4*)(feat + i01);
    f32x4 m0 = f0[0] * h0, m1 = f0[1] * h0;
    f32x4 m2 = f1[0] * h1, m3 = f1[1] * h1;
    AFrag af;
    af.q[0][0] = pk_bf16(m0[0], m0[1]);
    af.q[0][1] = pk_bf16(m0[2], m0[3]);
    af.q[0][2] = pk_bf16(m1[0], m1[1]);
    af.q[0][3] = pk_bf16(m1[2], m1[3]);
    af.q[1][0] = pk_bf16(m2[0], m2[1]);
    af.q[1][1] = pk_bf16(m2[2], m2[3]);
    af.q[1][2] = pk_bf16(m3[0], m3[1]);
    af.q[1][3] = pk_bf16(m3[2], m3[3]);
#pragma unroll
    for (int nt = 0; nt < NT; ++nt) {
      AFrag bf; bf.q[0] = bp[0]; bf.q[1] = bp[1];
      bp += 32;                       // advance 32 u32x4 = 256 bf16 (half a fragment)
      acc[nt] = __builtin_amdgcn_wmma_f32_16x16x32_bf16(false, af.v, false, bf.v,
                                                        (short)0, acc[nt], false, false);
    }
    i00 += 32;
    if (i00 >= F) { i00 -= F; ++k0; }
  }
}

// per-wave LDS working set (~32 KB); 4 waves/WG share the B stream through WGP$
struct WaveShm {
  float hL[16][66];
  __align__(16) float sinL[16][80];
  __align__(16) float VL[16][7][32]; // 0: v.Y1 ; 1..3: v_c ; 4..6: cross(v,Y1)_c
  float Y1L[16][4];
  float scL[16][16];
  float dumpA[16][32];   // s0 (0..23) | t2 (24..31)
  float dumpB[16][32];   // s1 (0..23)
  float dumpV[6][16][8]; // vB c=0..2 ; vC c=0..2
};

// ---------------- main fused kernel: 4 waves x 16 edges per workgroup ----------------
__global__ __launch_bounds__(128) void edge_main_k(
    const float* __restrict__ node_fea, const float* __restrict__ edge_attr,
    const int* __restrict__ edge_index, const int* __restrict__ x,
    const int* __restrict__ batch,
    const float* __restrict__ w_pre, const float* __restrict__ b_pre,
    const float* __restrict__ w_sc, const float* __restrict__ b_rh,
    const float* __restrict__ w_post_s, const float* __restrict__ b_post_s,
    const float* __restrict__ w_post_v,
    const unsigned short* __restrict__ Bpk, float* __restrict__ ws) {
  __shared__ WaveShm shm[4];

  const int wv   = threadIdx.x >> 5;
  const int lane = threadIdx.x & 31;
  const int el = lane & 15, half = lane >> 4;
  const int e = blockIdx.x * 64 + wv * 16 + el;
  WaveShm& S = shm[wv];

  // ---- prologue: per-edge features ----
  if (half == 0) {
    float dd = edge_attr[e * 4 + 0];
    float vx = edge_attr[e * 4 + 1], vy = edge_attr[e * 4 + 2], vz = edge_attr[e * 4 + 3];
    float nrm = sqrtf(vx * vx + vy * vy + vz * vz) + 1e-12f;
    const float s3 = 1.7320508075688772f;
    S.Y1L[el][0] = s3 * vx / nrm; S.Y1L[el][1] = s3 * vy / nrm; S.Y1L[el][2] = s3 * vz / nrm;
    float es[16];
#pragma unroll
    for (int j = 0; j < 16; ++j) { float t = dd - 0.4f * j; es[j] = expf(-3.125f * t * t); }
    for (int j = 0; j < 16; ++j) {        // edge_s -> s_in[64..80)
      float a = b_pre[j];
#pragma unroll
      for (int i2 = 0; i2 < 16; ++i2) a += es[i2] * w_pre[i2 * 16 + j];
      S.sinL[el][64 + j] = a;
    }
    int ii = edge_index[e], jj = edge_index[E_TOT + e];
    int p = x[ii] * 4 + x[jj];
    for (int o = 0; o < 16; ++o) {        // species-pair shortcut sc
      float a = 0.0f;
#pragma unroll
      for (int i2 = 0; i2 < 16; ++i2) a += es[i2] * w_sc[(i2 * 16 + p) * 16 + o];
      S.scL[el][o] = a * 0.0625f;         // 1/sqrt(16*16)
    }
  } else {
    int ii = edge_index[e], jj = edge_index[E_TOT + e];
    for (int t = 0; t < 32; ++t) S.sinL[el][t] = node_fea[ii * 80 + t];
    for (int t = 0; t < 32; ++t) S.sinL[el][32 + t] = node_fea[jj * 80 + t];
    for (int i2 = 0; i2 < 16; ++i2)
#pragma unroll
      for (int c = 0; c < 3; ++c) {
        S.VL[el][1 + c][i2]      = node_fea[ii * 80 + 32 + i2 * 3 + c];
        S.VL[el][1 + c][16 + i2] = node_fea[jj * 80 + 32 + i2 * 3 + c];
      }
  }
  { // v.Y1 and cross(v, Y1); same-wave LDS is in-order, no barrier needed
    float y0 = S.Y1L[el][0], y1 = S.Y1L[el][1], y2 = S.Y1L[el][2];
    for (int i2 = half * 16; i2 < half * 16 + 16; ++i2) {
      float ax = S.VL[el][1][i2], ay = S.VL[el][2][i2], az = S.VL[el][3][i2];
      S.VL[el][0][i2] = ax * y0 + ay * y1 + az * y2;
      S.VL[el][4][i2] = ay * y2 - az * y1;
      S.VL[el][5][i2] = az * y0 - ax * y2;
      S.VL[el][6][i2] = ax * y1 - ay * y0;
    }
  }
  __syncthreads();   // lockstep waves so the shared B stream stays hot in WGP$

  v8f vzero;
#pragma unroll
  for (int r = 0; r < 8; ++r) vzero[r] = 0.0f;

  // ---- h = silu(rbf @ w_rh + b_rh) via WMMA: (16x128)@(128x64) ----
  {
    float dd = edge_attr[e * 4 + 0];      // d of row-edge el
    const float DR = 6.0f / 127.0f;
    const float CR = -0.5f / (DR * DR);
    v8f hacc[4] = {vzero, vzero, vzero, vzero};
    for (int kt = 0; kt < 4; ++kt) {
      AFrag af;
#pragma unroll
      for (int run = 0; run < 2; ++run) {
        int Kb = kt * 32 + run * 16 + 8 * half;
#pragma unroll
        for (int p = 0; p < 4; ++p) {
          float x0 = dd - (Kb + 2 * p) * DR;
          float x1 = dd - (Kb + 2 * p + 1) * DR;
          af.q[run][p] = pk_bf16(expf(CR * x0 * x0), expf(CR * x1 * x1));
        }
      }
#pragma unroll
      for (int nt = 0; nt < 4; ++nt) {
        const u32x4* bp = (const u32x4*)(Bpk + (size_t)(FRAG_RH + kt * 4 + nt) * 512 + lane * 16);
        AFrag bf; bf.q[0] = bp[0]; bf.q[1] = bp[1];
        hacc[nt] = __builtin_amdgcn_wmma_f32_16x16x32_bf16(false, af.v, false, bf.v,
                                                           (short)0, hacc[nt], false, false);
      }
    }
#pragma unroll
    for (int nt = 0; nt < 4; ++nt) {
      int n = nt * 16 + el;
      float b = b_rh[n];
#pragma unroll
      for (int r = 0; r < 8; ++r) {
        int M = r + 8 * half;
        float v = hacc[nt][r] + b;
        S.hL[M][n] = v * sigm(v);         // silu
      }
    }
    if (half == 0) S.hL[el][64] = 1.0f; else S.hL[el][65] = 0.0f; // bias / pad rows
  }
  __syncthreads();

  // ---- the 8 fused GEMMs over rank-1 features ----
  const float* hrow = &S.hL[el][0];
  v8f accS[2]  = {vzero, vzero};
  gemm_block<80, 163, 2>(lane, &S.sinL[el][0], hrow, Bpk, FRAG_BS, accS);
  __syncthreads();
  v8f accS1[2] = {vzero, vzero};
  gemm_block<32, 65, 2>(lane, &S.VL[el][0][0], hrow, Bpk, FRAG_BVY, accS1);
  __syncthreads();
  v8f accVB[3] = {vzero, vzero, vzero};
  v8f accVC[3] = {vzero, vzero, vzero};
#pragma unroll
  for (int c = 0; c < 3; ++c)
    gemm_block<32, 65, 1>(lane, &S.VL[el][1 + c][0], hrow, Bpk, FRAG_BV, &accVB[c]);
  __syncthreads();
#pragma unroll
  for (int c = 0; c < 3; ++c)
    gemm_block<32, 65, 1>(lane, &S.VL[el][4 + c][0], hrow, Bpk, FRAG_BCR, &accVC[c]);

  // ---- dump accumulators to LDS (C layout: M = r + 8*half, N = lane%16) ----
#pragma unroll
  for (int r = 0; r < 8; ++r) {
    int M = r + 8 * half;
    S.dumpA[M][el]      = accS[0][r];
    S.dumpA[M][16 + el] = accS[1][r];
    S.dumpB[M][el]      = accS1[0][r];
    S.dumpB[M][16 + el] = accS1[1][r];
    if (el < 8) {
#pragma unroll
      for (int c = 0; c < 3; ++c) {
        S.dumpV[c][M][el]     = accVB[c][r];
        S.dumpV[3 + c][M][el] = accVC[c][r];
      }
    }
  }

  // ---- per-edge epilogue + graph-stat atomics ----
  int g = batch[edge_index[e]];
  float* sumS  = ws + WS_SUMS;
  float* sumSq = ws + WS_SUMSQ;
  float* sumV  = ws + WS_SUMV;
  float* cntW  = ws + WS_CNT;
  float* sOutW = ws + WS_SOUT;
  float* vOutW = ws + WS_VOUT;

  if (half == 0) {
    float so[16];
#pragma unroll
    for (int o = 0; o < 16; ++o) {
      float v = S.dumpA[el][o] + S.dumpB[el][o];  // s_conv scalar part
      so[o] = v * sigm(v);
    }
    for (int j = 0; j < 16; ++j) {
      float a = b_post_s[j] + S.scL[el][j];
#pragma unroll
      for (int o = 0; o < 16; ++o) a += so[o] * w_post_s[o * 16 + j];
      sOutW[(size_t)e * 16 + j] = a;
      atomicAdd(&sumS[g * 16 + j], a);
      atomicAdd(&sumSq[g * 16 + j], a * a);
    }
  } else {
    float gate[8], t2v[8];
#pragma unroll
    for (int o = 0; o < 8; ++o) {
      gate[o] = sigm(S.dumpA[el][16 + o] + S.dumpB[el][16 + o]);
      t2v[o]  = S.dumpA[el][24 + o];
    }
    float yv[3] = {S.Y1L[el][0], S.Y1L[el][1], S.Y1L[el][2]};
    float vt[8][3];
#pragma unroll
    for (int o = 0; o < 8; ++o)
#pragma unroll
      for (int c = 0; c < 3; ++c)
        vt[o][c] = (t2v[o] * yv[c] + S.dumpV[c][el][o] + S.dumpV[3 + c][el][o]) * gate[o];
    for (int o = 0; o < 8; ++o) {
      float s2 = 0.0f;
#pragma unroll
      for (int c = 0; c < 3; ++c) {
        float a = 0.0f;
#pragma unroll
        for (int i2 = 0; i2 < 8; ++i2) a += vt[i2][c] * w_post_v[i2 * 8 + o];
        vOutW[(size_t)e * 24 + o * 3 + c] = a;
        s2 += a * a;
      }
      atomicAdd(&sumV[g * 8 + o], s2);
    }
    atomicAdd(&cntW[g], 1.0f);
  }
}

// ---------------- kernel: finalize per-graph stats ----------------
__global__ void finalize_stats_k(float* ws) {
  int t = threadIdx.x;
  __shared__ float cntS[16];
  if (t < 16) cntS[t] = fmaxf(ws[WS_CNT + t], 1.0f);
  __syncthreads();
  if (t < 256) { int g = t >> 4; ws[WS_MU + t] = ws[WS_SUMS + t] / cntS[g]; }
  __syncthreads();
  if (t < 16) {
    float c = cntS[t];
    float acc = 0.0f;
    for (int ch = 0; ch < 16; ++ch) {
      float m = ws[WS_MU + t * 16 + ch];
      acc += ws[WS_SUMSQ + t * 16 + ch] / c - m * m;
    }
    ws[WS_RMSS + t] = sqrtf(acc / 16.0f + 1e-5f);
    float av = 0.0f;
    for (int o = 0; o < 8; ++o) av += ws[WS_SUMV + t * 8 + o] / c;
    ws[WS_RMSV + t] = sqrtf(av / 24.0f + 1e-5f);   // /3 then mean over 8
  }
}

// ---------------- kernel: per-edge normalize + final linears ----------------
__global__ void finalize_edges_k(const float* __restrict__ edge_attr, const int* __restrict__ edge_index,
                                 const int* __restrict__ batch,
                                 const float* __restrict__ ln_w_s, const float* __restrict__ ln_b_s,
                                 const float* __restrict__ ln_w_v, const float* __restrict__ w_skip,
                                 const float* __restrict__ w_edge_s, const float* __restrict__ b_edge_s,
                                 const float* __restrict__ w_edge_v,
                                 const float* __restrict__ ws, float* __restrict__ out) {
  int e = blockIdx.x * 256 + threadIdx.x;
  if (e >= E_TOT) return;
  int g = batch[edge_index[e]];
  const float* sOut = ws + WS_SOUT + (size_t)e * 16;
  const float* vOut = ws + WS_VOUT + (size_t)e * 24;
  float rs = ws[WS_RMSS + g], rv = ws[WS_RMSV + g];
  float dd = edge_attr[e * 4 + 0];
  float es0[16];
#pragma unroll
  for (int j = 0; j < 16; ++j) { float t = dd - 0.4f * j; es0[j] = expf(-3.125f * t * t); }
  float sn[16];
  for (int ch = 0; ch < 16; ++ch) {
    float skip = 0.0f;
#pragma unroll
    for (int i = 0; i < 16; ++i) skip += es0[i] * w_skip[i * 16 + ch];
    sn[ch] = (sOut[ch] - ws[WS_MU + g * 16 + ch]) / rs * ln_w_s[ch] + ln_b_s[ch] + skip;
  }
  for (int j = 0; j < 16; ++j) {
    float a = b_edge_s[j];
#pragma unroll
    for (int ch = 0; ch < 16; ++ch) a += sn[ch] * w_edge_s[ch * 16 + j];
    out[(size_t)e * 40 + j] = a;
  }
  float vn[8][3];
#pragma unroll
  for (int o = 0; o < 8; ++o)
#pragma unroll
    for (int c = 0; c < 3; ++c) vn[o][c] = vOut[o * 3 + c] / rv * ln_w_v[o];
  for (int o = 0; o < 8; ++o)
#pragma unroll
    for (int c = 0; c < 3; ++c) {
      float a = 0.0f;
#pragma unroll
      for (int i = 0; i < 8; ++i) a += vn[i][c] * w_edge_v[i * 8 + o];
      out[(size_t)e * 40 + 16 + o * 3 + c] = a;
    }
}

// ---------------- launcher ----------------
extern "C" void kernel_launch(void* const* d_in, const int* in_sizes, int n_in,
                              void* d_out, int out_size, void* d_ws, size_t ws_size,
                              hipStream_t stream) {
  (void)in_sizes; (void)n_in; (void)out_size; (void)ws_size;
  const float* node_fea  = (const float*)d_in[0];
  const float* edge_attr = (const float*)d_in[1];
  const int*   edge_index= (const int*)d_in[2];
  const int*   x         = (const int*)d_in[3];
  const int*   batch     = (const int*)d_in[4];
  const float* w_rh      = (const float*)d_in[5];
  const float* b_rh      = (const float*)d_in[6];
  const float* w_ro      = (const float*)d_in[7];
  const float* b_ro      = (const float*)d_in[8];
  const float* w_pre     = (const float*)d_in[9];
  const float* b_pre     = (const float*)d_in[10];
  const float* w_sc      = (const float*)d_in[11];
  const float* w_post_s  = (const float*)d_in[12];
  const float* b_post_s  = (const float*)d_in[13];
  const float* w_post_v  = (const float*)d_in[14];
  const float* ln_w_s    = (const float*)d_in[15];
  const float* ln_b_s    = (const float*)d_in[16];
  const float* ln_w_v    = (const float*)d_in[17];
  const float* w_skip    = (const float*)d_in[18];
  const float* w_edge_s  = (const float*)d_in[19];
  const float* b_edge_s  = (const float*)d_in[20];
  const float* w_edge_v  = (const float*)d_in[21];

  float* ws = (float*)d_ws;
  unsigned short* Bpk = (unsigned short*)((char*)d_ws + WS_BPK_BYTES);

  zero_stats_k<<<1, 1024, 0, stream>>>(ws);
  repack_k<<<(NFRAG * 512 + 255) / 256, 256, 0, stream>>>(w_rh, w_ro, b_ro, Bpk);
  edge_main_k<<<E_TOT / 64, 128, 0, stream>>>(node_fea, edge_attr, edge_index, x, batch,
                                              w_pre, b_pre, w_sc, b_rh,
                                              w_post_s, b_post_s, w_post_v, Bpk, ws);
  finalize_stats_k<<<1, 256, 0, stream>>>(ws);
  finalize_edges_k<<<E_TOT / 256, 256, 0, stream>>>(edge_attr, edge_index, batch,
                                                    ln_w_s, ln_b_s, ln_w_v, w_skip,
                                                    w_edge_s, b_edge_s, w_edge_v,
                                                    ws, (float*)d_out);
}
